// TKAN_87651692577430
// MI455X (gfx1250) — compile-verified, matrix operand order checked by reading, not run on hardware
//
#include <hip/hip_runtime.h>

// ---------------------------------------------------------------------------
// TKAN on MI455X (gfx1250): bf16 WMMA GEMMs + persistent recurrent kernel.
// Shapes: B=64, T=256, D=512, H=1024, n=3, d=512.
// ---------------------------------------------------------------------------

typedef __bf16 bf16_t;
typedef __attribute__((ext_vector_type(16))) __bf16 v16bf;
typedef __attribute__((ext_vector_type(8)))  __bf16 v8bf;
typedef __attribute__((ext_vector_type(8)))  float  v8f;

static constexpr int Bb = 64, Tt = 256, Dd = 512, Hh = 1024, NN = 3, dd = 512;
static constexpr int NCAT = 3 * Hh + NN * dd;       // 4608 (gates | 3 x-branches)
static constexpr int ROWS = Bb * Tt;                // 16384
static constexpr int NBLK = 64;                     // persistent grid
static constexpr int NWAVE = NBLK * 8;              // 512 wave32 waves

// ---- workspace layout (bytes) ----
static constexpr size_t OFF_XB    = 0;                          // bf16 x        [16384,512]
static constexpr size_t OFF_WCAT  = OFF_XB    + (size_t)ROWS*Dd*2;          // bf16 [4608][512] N-major
static constexpr size_t OFF_WHHT  = OFF_WCAT  + (size_t)NCAT*Dd*2;          // bf16 [3072][1024]
static constexpr size_t OFF_KHT   = OFF_WHHT  + (size_t)3*Hh*Hh*2;          // bf16 [3][512][512]
static constexpr size_t OFF_LWT   = OFF_KHT   + (size_t)NN*dd*dd*2;         // bf16 [3][512][512]
static constexpr size_t OFF_WAGGT = OFF_LWT   + (size_t)NN*dd*dd*2;         // bf16 [1024][1536]
static constexpr size_t OFF_XPRE  = OFF_WAGGT + (size_t)Hh*(NN*dd)*2;       // bf16 [16384][4608]
static constexpr size_t OFF_HB    = OFF_XPRE  + (size_t)ROWS*NCAT*2;        // bf16 [64][1024]
static constexpr size_t OFF_C     = OFF_HB    + (size_t)Bb*Hh*2;            // f32  [64][1024]
static constexpr size_t OFF_TC    = OFF_C     + (size_t)Bb*Hh*4;            // f32  [64][1024]
static constexpr size_t OFF_SUB   = OFF_TC    + (size_t)Bb*Hh*4;            // f32  [3][64][512]
static constexpr size_t OFF_SUBB  = OFF_SUB   + (size_t)NN*Bb*dd*4;         // bf16 [3][64][512]
static constexpr size_t OFF_GBUF  = OFF_SUBB  + (size_t)NN*Bb*dd*2;         // f32  [64][3072]
static constexpr size_t OFF_ABUFB = OFF_GBUF  + (size_t)Bb*3*Hh*4;          // bf16 [3][64][512]
static constexpr size_t OFF_ZBUF  = OFF_ABUFB + (size_t)NN*Bb*dd*2;         // f32  [3][64][512]
static constexpr size_t OFF_AGGB  = OFF_ZBUF  + (size_t)NN*Bb*dd*4;         // bf16 [64][1536]
static constexpr size_t OFF_SYNC  = OFF_AGGB  + (size_t)Bb*NN*dd*2;         // u32  sync words
static constexpr size_t WS_NEEDED = OFF_SYNC + 256;

// ---------------------------------------------------------------------------
// WMMA fragment helpers (bf16 16x16x32, f32 accumulate).
// ---------------------------------------------------------------------------
__device__ __forceinline__ v16bf load_frag(const bf16_t* base, int ld) {
  const int lane = threadIdx.x & 31;
  const int r  = lane & 15;
  const int kb = (lane & 16) ? 8 : 0;
  const bf16_t* p = base + r * ld + kb;
  v8bf lo = *(const v8bf*)(p);
  v8bf hi = *(const v8bf*)(p + 16);
  return __builtin_shufflevector(lo, hi, 0,1,2,3,4,5,6,7,8,9,10,11,12,13,14,15);
}

__device__ __forceinline__ v8f wmma_bf16(v16bf a, v16bf b, v8f c) {
  return __builtin_amdgcn_wmma_f32_16x16x32_bf16(false, a, false, b, (short)0, c, false, false);
}

__device__ __forceinline__ void store_tile_f32(float* base, int ld, v8f c) {
  const int lane = threadIdx.x & 31;
  const int col = lane & 15;
  const int rb  = (lane & 16) ? 8 : 0;
#pragma unroll
  for (int v = 0; v < 8; ++v) base[(size_t)(rb + v) * ld + col] = c[v];
}

__device__ __forceinline__ void store_tile_bf16(bf16_t* base, int ld, v8f c) {
  const int lane = threadIdx.x & 31;
  const int col = lane & 15;
  const int rb  = (lane & 16) ? 8 : 0;
#pragma unroll
  for (int v = 0; v < 8; ++v) base[(size_t)(rb + v) * ld + col] = (bf16_t)c[v];
}

__device__ __forceinline__ v8f load_ctile_bf16(const bf16_t* base, int ld) {
  const int lane = threadIdx.x & 31;
  const int col = lane & 15;
  const int rb  = (lane & 16) ? 8 : 0;
  v8f c;
#pragma unroll
  for (int v = 0; v < 8; ++v) c[v] = (float)base[(size_t)(rb + v) * ld + col];
  return c;
}

__device__ __forceinline__ v8f cinit_bcast(const float* colbias) {
  const float b = colbias[threadIdx.x & 15];
  v8f c;
#pragma unroll
  for (int v = 0; v < 8; ++v) c[v] = b;
  return c;
}

__device__ __forceinline__ float sigmoidf_(float x) { return 1.0f / (1.0f + __expf(-x)); }

// Device-wide arrive/wait barrier (persistent kernel, fixed 64 blocks).
__device__ __forceinline__ void grid_sync(unsigned* counter, unsigned* gen, unsigned nblk) {
  __syncthreads();
  if (threadIdx.x == 0) {
    __threadfence();
    unsigned g = __atomic_load_n(gen, __ATOMIC_RELAXED);
    unsigned prev = __atomic_fetch_add(counter, 1u, __ATOMIC_ACQ_REL);
    if (prev == nblk - 1) {
      __atomic_store_n(counter, 0u, __ATOMIC_RELAXED);
      __atomic_fetch_add(gen, 1u, __ATOMIC_RELEASE);
    } else {
      while (__atomic_load_n(gen, __ATOMIC_ACQUIRE) == g) __builtin_amdgcn_s_sleep(2);
    }
    __threadfence();
  }
  __syncthreads();
}

// ---------------------------------------------------------------------------
// Prep kernels
// ---------------------------------------------------------------------------
__global__ void tkan_zero_u32(unsigned* p, int n) {
  int i = blockIdx.x * blockDim.x + threadIdx.x;
  if (i < n) p[i] = 0u;
}

__global__ void tkan_cast_bf16(const float* __restrict__ s, bf16_t* __restrict__ d, int n) {
  int i = blockIdx.x * blockDim.x + threadIdx.x;
  if (i < n) d[i] = (bf16_t)s[i];
}

// dst[n*rows + k] = src[k*cols + n]  (store weight matrices N-major for B frags)
__global__ void tkan_transpose_cast(const float* __restrict__ s, bf16_t* __restrict__ d,
                                    int rows, int cols) {
  int i = blockIdx.x * blockDim.x + threadIdx.x;
  if (i < rows * cols) {
    int k = i / cols, n = i % cols;
    d[(size_t)n * rows + k] = (bf16_t)s[i];
  }
}

// ---------------------------------------------------------------------------
// Phase 1: Xpre[16384,4608] = x_bf16 @ [W_ih | K_x0 | K_x1 | K_x2]   (bf16 out)
// 4x2 register blocking: 8 WMMAs per 6 fragment loads (~21 FLOP/B from L2).
// ---------------------------------------------------------------------------
__global__ __launch_bounds__(256)
void tkan_gemm_xpre(const bf16_t* __restrict__ xb, const bf16_t* __restrict__ Wcat,
                    bf16_t* __restrict__ Xpre) {
  const int wave = (blockIdx.x * blockDim.x + threadIdx.x) >> 5;
  const int nw   = (gridDim.x * blockDim.x) >> 5;
  const int MT4 = ROWS / 64;                 // 256 macro rows (4 M-tiles each)
  const int NT2 = NCAT / 32;                 // 144 macro cols (2 N-tiles each)
  for (int ti = wave; ti < MT4 * NT2; ti += nw) {
    const int tm4 = ti / NT2, tn2 = ti % NT2;
    const bf16_t* A0 = xb   + (size_t)tm4 * 64 * Dd;
    const bf16_t* B0 = Wcat + (size_t)tn2 * 32 * Dd;
    v8f acc[8] = {};                         // acc[mi*2+ni]
    for (int k = 0; k < Dd; k += 32) {
      __builtin_prefetch(A0 + k + 256, 0, 1);
      __builtin_prefetch(B0 + k + 256, 0, 1);
      v16bf bf0 = load_frag(B0 + k, Dd);
      v16bf bf1 = load_frag(B0 + 16 * Dd + k, Dd);
#pragma unroll
      for (int mi = 0; mi < 4; ++mi) {
        v16bf af = load_frag(A0 + (size_t)mi * 16 * Dd + k, Dd);
        acc[mi * 2 + 0] = wmma_bf16(af, bf0, acc[mi * 2 + 0]);
        acc[mi * 2 + 1] = wmma_bf16(af, bf1, acc[mi * 2 + 1]);
      }
    }
#pragma unroll
    for (int mi = 0; mi < 4; ++mi)
#pragma unroll
      for (int ni = 0; ni < 2; ++ni)
        store_tile_bf16(Xpre + ((size_t)tm4 * 64 + mi * 16) * NCAT + tn2 * 32 + ni * 16,
                        NCAT, acc[mi * 2 + ni]);
  }
}

// ---------------------------------------------------------------------------
// Phase 2: persistent recurrent kernel. 64 blocks x 256 thr, 4 stages/step.
// Recurrent GEMMs use 2x1 M-blocking so each weight (B) fragment feeds 2 WMMAs.
// ---------------------------------------------------------------------------
struct RecArgs {
  const bf16_t *Xpre, *WhhT, *KhT, *LwT, *WaggT;
  bf16_t *h_b, *sub_b, *abuf_b, *aggb;
  float *c, *tc, *sub, *gbuf, *zbuf;
  const float *bias, *lin_b, *ln_g, *ln_b, *sub_kernel, *b_agg;
  float *out;
  unsigned *sync;
};

__global__ __launch_bounds__(256)
void tkan_recurrent(RecArgs a) {
  const int tid  = threadIdx.x;
  const int lane = tid & 31;
  const int wave = (blockIdx.x * blockDim.x + tid) >> 5;   // 0..511
  unsigned* cnt = a.sync;
  unsigned* gen = a.sync + 16;
  __shared__ float red[256];
  __shared__ float zsh[3 * 512];                           // staged LN rows (6 KB)

  for (int t = 0; t < Tt; ++t) {
    // ---- Stage 1: gbuf = Xp(t) + h@W_hh ; abuf = Xk(t) + sub@K_h
    //      576 strips of 2 M-tiles each (384 gate strips + 192 branch strips).
    for (int s = wave; s < 576; s += NWAVE) {
      if (s < 384) {                                       // gates: K=1024
        const int tmp = s / 192, tn = s % 192;
        const int m0 = tmp * 32;
        v8f acc0 = load_ctile_bf16(a.Xpre + ((size_t)m0 * Tt + t) * NCAT + tn * 16, Tt * NCAT);
        v8f acc1 = load_ctile_bf16(a.Xpre + ((size_t)(m0 + 16) * Tt + t) * NCAT + tn * 16,
                                   Tt * NCAT);
        const bf16_t* Bp = a.WhhT + (size_t)tn * 16 * Hh;
        for (int k = 0; k < Hh; k += 32) {
          v16bf bf = load_frag(Bp + k, Hh);
          acc0 = wmma_bf16(load_frag(a.h_b + (size_t)m0 * Hh + k, Hh), bf, acc0);
          acc1 = wmma_bf16(load_frag(a.h_b + (size_t)(m0 + 16) * Hh + k, Hh), bf, acc1);
        }
        store_tile_f32(a.gbuf + (size_t)m0 * (3 * Hh) + tn * 16, 3 * Hh, acc0);
        store_tile_f32(a.gbuf + (size_t)(m0 + 16) * (3 * Hh) + tn * 16, 3 * Hh, acc1);
      } else {                                             // agg_in: K=512
        const int u = s - 384;
        const int n = u / 64, rem = u % 64, tmp = rem / 32, tn = rem % 32;
        const int m0 = tmp * 32;
        const size_t xc = 3 * Hh + (size_t)n * dd + tn * 16;
        v8f acc0 = load_ctile_bf16(a.Xpre + ((size_t)m0 * Tt + t) * NCAT + xc, Tt * NCAT);
        v8f acc1 = load_ctile_bf16(a.Xpre + ((size_t)(m0 + 16) * Tt + t) * NCAT + xc, Tt * NCAT);
        const bf16_t* A  = a.sub_b + (size_t)n * Bb * dd;
        const bf16_t* Bp = a.KhT   + (size_t)n * dd * dd + (size_t)tn * 16 * dd;
        for (int k = 0; k < dd; k += 32) {
          v16bf bf = load_frag(Bp + k, dd);
          acc0 = wmma_bf16(load_frag(A + (size_t)m0 * dd + k, dd), bf, acc0);
          acc1 = wmma_bf16(load_frag(A + (size_t)(m0 + 16) * dd + k, dd), bf, acc1);
        }
        bf16_t* O = a.abuf_b + (size_t)n * Bb * dd + tn * 16;
        store_tile_bf16(O + (size_t)m0 * dd, dd, acc0);
        store_tile_bf16(O + (size_t)(m0 + 16) * dd, dd, acc1);
      }
    }
    grid_sync(cnt, gen, NBLK);

    // ---- Stage 2a: LSTM cell update (elementwise, 64x1024)
    for (int e = blockIdx.x * 256 + tid; e < Bb * Hh; e += NBLK * 256) {
      const int b = e >> 10, j = e & (Hh - 1);
      const float gi = sigmoidf_(a.gbuf[(size_t)b * 3 * Hh + j]          + a.bias[j]);
      const float gf = sigmoidf_(a.gbuf[(size_t)b * 3 * Hh + Hh + j]     + a.bias[Hh + j]);
      const float gc = sigmoidf_(a.gbuf[(size_t)b * 3 * Hh + 2 * Hh + j] + a.bias[2 * Hh + j]);
      const float cn = gf * a.c[e] + gi * tanhf(gc);
      a.c[e]  = cn;
      a.tc[e] = tanhf(cn);
    }
    // ---- Stage 2b: z = agg_in @ lin_w + lin_b  (192 strips, K=512)
    for (int s = wave; s < 192; s += NWAVE) {
      const int n = s / 64, rem = s % 64, tmp = rem / 32, tn = rem % 32;
      const int m0 = tmp * 32;
      v8f acc0 = cinit_bcast(a.lin_b + n * dd + tn * 16);
      v8f acc1 = acc0;
      const bf16_t* A  = a.abuf_b + (size_t)n * Bb * dd;
      const bf16_t* Bp = a.LwT    + (size_t)n * dd * dd + (size_t)tn * 16 * dd;
      for (int k = 0; k < dd; k += 32) {
        v16bf bf = load_frag(Bp + k, dd);
        acc0 = wmma_bf16(load_frag(A + (size_t)m0 * dd + k, dd), bf, acc0);
        acc1 = wmma_bf16(load_frag(A + (size_t)(m0 + 16) * dd + k, dd), bf, acc1);
      }
      float* O = a.zbuf + (size_t)n * Bb * dd + tn * 16;
      store_tile_f32(O + (size_t)m0 * dd, dd, acc0);
      store_tile_f32(O + (size_t)(m0 + 16) * dd, dd, acc1);
    }
    grid_sync(cnt, gen, NBLK);

    // ---- Stage 3: LayerNorm(z) -> aggb (bf16), new_sub = kh*y + kx*sub.
    // Stage this block's 3 z-rows (6 KB) into LDS with async b128 copies.
    for (int i = tid; i < 384; i += 256) {
      const int rr  = i >> 7;                              // row within block (0..2)
      const int row = blockIdx.x * 3 + rr;                 // 0..191
      const int n = row >> 6, b = row & 63;
      const unsigned long long g =
          (unsigned long long)(const void*)((const char*)(a.zbuf + (size_t)n * Bb * dd +
                                                          (size_t)b * dd) + (i & 127) * 16);
      const unsigned ldsoff = (unsigned)(unsigned long long)(const void*)zsh + (unsigned)(i * 16);
      asm volatile("global_load_async_to_lds_b128 %0, %1, off"
                   :: "v"(ldsoff), "v"(g) : "memory");
    }
    asm volatile("s_wait_asynccnt 0x0" ::: "memory");
    __syncthreads();
    for (int r = 0; r < 3; ++r) {
      const int row = blockIdx.x * 3 + r;                  // 0..191
      const int n = row >> 6, b = row & 63;
      const float* z = zsh + r * 512;
      const float s0 = z[tid], s1 = z[tid + 256];
      red[tid] = s0 + s1; __syncthreads();
      for (int off = 128; off > 0; off >>= 1) {
        if (tid < off) red[tid] += red[tid + off];
        __syncthreads();
      }
      const float mu = red[0] * (1.0f / (float)dd); __syncthreads();
      red[tid] = s0 * s0 + s1 * s1; __syncthreads();
      for (int off = 128; off > 0; off >>= 1) {
        if (tid < off) red[tid] += red[tid + off];
        __syncthreads();
      }
      const float var = red[0] * (1.0f / (float)dd) - mu * mu; __syncthreads();
      const float rs = rsqrtf(var + 1e-5f);
#pragma unroll
      for (int e = 0; e < 2; ++e) {
        const int j = tid + e * 256;
        const float y = (z[j] - mu) * rs * a.ln_g[n * dd + j] + a.ln_b[n * dd + j];
        a.aggb[(size_t)b * (NN * dd) + n * dd + j] = (bf16_t)y;
        const size_t si = (size_t)n * Bb * dd + (size_t)b * dd + j;
        const float ns = a.sub_kernel[n * 2 * dd + j] * y
                       + a.sub_kernel[n * 2 * dd + dd + j] * a.sub[si];
        a.sub[si]   = ns;
        a.sub_b[si] = (bf16_t)ns;
      }
      __syncthreads();
    }
    grid_sync(cnt, gen, NBLK);

    // ---- Stage 4: o = sigmoid(aggb @ W_agg + b_agg); h = o*tanh(c)  (128 strips, K=1536)
    for (int s = wave; s < 128; s += NWAVE) {
      const int tmp = s >> 6, tn = s & 63;
      const int m0 = tmp * 32;
      v8f acc0 = cinit_bcast(a.b_agg + tn * 16);
      v8f acc1 = acc0;
      const bf16_t* Bp = a.WaggT + (size_t)tn * 16 * (NN * dd);
      for (int k = 0; k < NN * dd; k += 32) {
        v16bf bf = load_frag(Bp + k, NN * dd);
        acc0 = wmma_bf16(load_frag(a.aggb + (size_t)m0 * (NN * dd) + k, NN * dd), bf, acc0);
        acc1 = wmma_bf16(load_frag(a.aggb + (size_t)(m0 + 16) * (NN * dd) + k, NN * dd), bf, acc1);
      }
      const int col = lane & 15, rb = (lane & 16) ? 8 : 0;
#pragma unroll
      for (int v = 0; v < 8; ++v) {
        {
          const int b = m0 + rb + v, j = tn * 16 + col;
          const float h = sigmoidf_(acc0[v]) * a.tc[(size_t)b * Hh + j];
          a.out[((size_t)b * Tt + t) * Hh + j] = h;
          a.h_b[(size_t)b * Hh + j] = (bf16_t)h;
        }
        {
          const int b = m0 + 16 + rb + v, j = tn * 16 + col;
          const float h = sigmoidf_(acc1[v]) * a.tc[(size_t)b * Hh + j];
          a.out[((size_t)b * Tt + t) * Hh + j] = h;
          a.h_b[(size_t)b * Hh + j] = (bf16_t)h;
        }
      }
    }
    grid_sync(cnt, gen, NBLK);
  }
}

// ---------------------------------------------------------------------------
extern "C" void kernel_launch(void* const* d_in, const int* in_sizes, int n_in,
                              void* d_out, int out_size, void* d_ws, size_t ws_size,
                              hipStream_t stream) {
  if (ws_size < WS_NEEDED) return;  // scratch too small: bail deterministically

  const float* x      = (const float*)d_in[0];
  const float* h0     = (const float*)d_in[1];
  const float* c0     = (const float*)d_in[2];
  const float* sub0   = (const float*)d_in[3];
  const float* W_ih   = (const float*)d_in[4];
  const float* W_hh   = (const float*)d_in[5];
  const float* bias   = (const float*)d_in[6];
  const float* K_x    = (const float*)d_in[7];
  const float* K_h    = (const float*)d_in[8];
  const float* subk   = (const float*)d_in[9];
  const float* lin_w  = (const float*)d_in[10];
  const float* lin_b  = (const float*)d_in[11];
  const float* ln_g   = (const float*)d_in[12];
  const float* ln_b   = (const float*)d_in[13];
  const float* W_agg  = (const float*)d_in[14];
  const float* b_agg  = (const float*)d_in[15];

  char* ws = (char*)d_ws;
  bf16_t*   xb     = (bf16_t*)(ws + OFF_XB);
  bf16_t*   Wcat   = (bf16_t*)(ws + OFF_WCAT);
  bf16_t*   WhhT   = (bf16_t*)(ws + OFF_WHHT);
  bf16_t*   KhT    = (bf16_t*)(ws + OFF_KHT);
  bf16_t*   LwT    = (bf16_t*)(ws + OFF_LWT);
  bf16_t*   WaggT  = (bf16_t*)(ws + OFF_WAGGT);
  bf16_t*   Xpre   = (bf16_t*)(ws + OFF_XPRE);
  bf16_t*   h_b    = (bf16_t*)(ws + OFF_HB);
  float*    c_st   = (float*)(ws + OFF_C);
  float*    tc_st  = (float*)(ws + OFF_TC);
  float*    sub_st = (float*)(ws + OFF_SUB);
  bf16_t*   sub_b  = (bf16_t*)(ws + OFF_SUBB);
  float*    gbuf   = (float*)(ws + OFF_GBUF);
  bf16_t*   abuf_b = (bf16_t*)(ws + OFF_ABUFB);
  float*    zbuf   = (float*)(ws + OFF_ZBUF);
  bf16_t*   aggb   = (bf16_t*)(ws + OFF_AGGB);
  unsigned* sync   = (unsigned*)(ws + OFF_SYNC);

  auto nb = [](int n) { return (n + 255) / 256; };

  // --- prep: barrier state, bf16 casts, N-major weight transposes ---
  tkan_zero_u32<<<1, 64, 0, stream>>>(sync, 64);
  tkan_cast_bf16<<<nb(ROWS * Dd), 256, 0, stream>>>(x, xb, ROWS * Dd);
  tkan_transpose_cast<<<nb(Dd * 3 * Hh), 256, 0, stream>>>(W_ih, Wcat, Dd, 3 * Hh);
  for (int n = 0; n < NN; ++n)
    tkan_transpose_cast<<<nb(Dd * dd), 256, 0, stream>>>(
        K_x + (size_t)n * Dd * dd, Wcat + (size_t)(3 * Hh + n * dd) * Dd, Dd, dd);
  tkan_transpose_cast<<<nb(Hh * 3 * Hh), 256, 0, stream>>>(W_hh, WhhT, Hh, 3 * Hh);
  for (int n = 0; n < NN; ++n)
    tkan_transpose_cast<<<nb(dd * dd), 256, 0, stream>>>(
        K_h + (size_t)n * dd * dd, KhT + (size_t)n * dd * dd, dd, dd);
  for (int n = 0; n < NN; ++n)
    tkan_transpose_cast<<<nb(dd * dd), 256, 0, stream>>>(
        lin_w + (size_t)n * dd * dd, LwT + (size_t)n * dd * dd, dd, dd);
  tkan_transpose_cast<<<nb(NN * dd * Hh), 256, 0, stream>>>(W_agg, WaggT, NN * dd, Hh);
  tkan_cast_bf16<<<nb(Bb * Hh), 256, 0, stream>>>(h0, h_b, Bb * Hh);
  tkan_cast_bf16<<<nb(NN * Bb * dd), 256, 0, stream>>>(sub0, sub_b, NN * Bb * dd);
  hipMemcpyAsync(c_st, c0, (size_t)Bb * Hh * sizeof(float), hipMemcpyDeviceToDevice, stream);
  hipMemcpyAsync(sub_st, sub0, (size_t)NN * Bb * dd * sizeof(float), hipMemcpyDeviceToDevice,
                 stream);

  // --- phase 1: big parallel GEMM over all timesteps ---
  tkan_gemm_xpre<<<1024, 256, 0, stream>>>(xb, Wcat, Xpre);

  // --- phase 2: persistent recurrence ---
  RecArgs ra;
  ra.Xpre = Xpre; ra.WhhT = WhhT; ra.KhT = KhT; ra.LwT = LwT; ra.WaggT = WaggT;
  ra.h_b = h_b; ra.sub_b = sub_b; ra.abuf_b = abuf_b; ra.aggb = aggb;
  ra.c = c_st; ra.tc = tc_st; ra.sub = sub_st; ra.gbuf = gbuf; ra.zbuf = zbuf;
  ra.bias = bias; ra.lin_b = lin_b; ra.ln_g = ln_g; ra.ln_b = ln_b;
  ra.sub_kernel = subk; ra.b_agg = b_agg;
  ra.out = (float*)d_out; ra.sync = sync;
  tkan_recurrent<<<NBLK, 256, 0, stream>>>(ra);
}
